// PhaseAttention_77618648973383
// MI455X (gfx1250) — matrix operand order repaired and checked by reading, not compiled, not run on hardware
//
#include <hip/hip_runtime.h>
#include <stdint.h>

// ---------------- problem constants ----------------
static constexpr int B  = 4;
static constexpr int S  = 2048;
static constexpr int H  = 1024;
static constexpr int NF = H / 2 + 1;      // 513 rfft frequencies
static constexpr int FP = 528;            // NF padded to multiple of 16
static constexpr int KK = 2 * FP;         // [cos | sin] feature dim = 1056
static constexpr int WPB = 8;             // waves per 256-thread block (wave32)

typedef __attribute__((ext_vector_type(16))) __bf16 v16bf;
typedef __attribute__((ext_vector_type(8)))  float  v8f;

union Frag16 {
    v16bf v;
    __bf16 h[16];
    uint4  q[2];
};

#define WMMA_BF16(a, b, c) \
    __builtin_amdgcn_wmma_f32_16x16x32_bf16(false, (a), false, (b), (short)0, (c), false, false)

__device__ __forceinline__ unsigned short bfbits(float x) {
    __bf16 b = (__bf16)x;                 // native v_cvt to bf16
    return __builtin_bit_cast(unsigned short, b);
}

// A fragment (16x32 bf16) from a bf16 row: two contiguous 16B runs.
// rowBase must already include "+ hi*16" lane offset. K(e) = e + 16*hi.
__device__ __forceinline__ v16bf load_a_bf16(const unsigned short* rowBase) {
    Frag16 a;
    a.q[0] = *(const uint4*)(rowBase);        // e = 0..7
    a.q[1] = *(const uint4*)(rowBase + 8);    // e = 8..15
    return a.v;
}

// B fragment (32x16 bf16) from a K-contiguous bf16 row (operand stored
// transposed so per-lane K runs are contiguous): one 32B run.
// p must already include "+ hi*16" lane offset. K(e) = e + 16*hi, N = lane&15.
__device__ __forceinline__ v16bf load_b_bf16(const unsigned short* p) {
    Frag16 b;
    b.q[0] = *(const uint4*)(p);
    b.q[1] = *(const uint4*)(p + 8);
    return b.v;
}

// ---------------------------------------------------------------
// Kernel 0: bf16 DFT twiddle basis, stored TRANSPOSED [FP][H] so the
// GEMM B-fragments read contiguously along K(=h). Exact integer arg
// reduction: (h*f) & 1023 keeps sincos args in [0, 2pi).
// ---------------------------------------------------------------
__global__ void pa_basis_kernel(unsigned short* __restrict__ cosBT,
                                unsigned short* __restrict__ sinBT) {
    int idx = blockIdx.x * blockDim.x + threadIdx.x;
    if (idx >= FP * H) return;
    int f = idx / H, h = idx % H;
    unsigned short c = 0, s = 0;
    if (f < NF) {
        int m = (h * f) & (H - 1);
        float ang = (float)m * (6.28318530717958647692f / (float)H);
        c = bfbits(__cosf(ang));
        s = bfbits(__sinf(ang));
    }
    cosBT[idx] = c; sinBT[idx] = s;
}

// ---------------------------------------------------------------
// Kernel 1: one-shot f32 -> bf16 convert of Q and K (A-side operands
// are reused by 33 freq-tiles each; convert once, not per k-step).
// 8 elements per thread: 2x b128 load, 4x cvt_pk, 1x b128 store.
// ---------------------------------------------------------------
__global__ void pa_cvt_kernel(const float* __restrict__ Q,
                              const float* __restrict__ K,
                              unsigned short* __restrict__ Qh,
                              unsigned short* __restrict__ Kh) {
    const size_t n = (size_t)B * S * H;
    size_t i = ((size_t)blockIdx.x * 256 + threadIdx.x) * 8;
    const float* src;
    unsigned short* dst;
    if (i < n) { src = Q + i;       dst = Qh + i; }
    else       { src = K + (i - n); dst = Kh + (i - n); }
    float4 f0 = ((const float4*)src)[0];
    float4 f1 = ((const float4*)src)[1];
    union { uint4 q; __bf16 h[8]; } o;
    o.h[0]=(__bf16)f0.x; o.h[1]=(__bf16)f0.y; o.h[2]=(__bf16)f0.z; o.h[3]=(__bf16)f0.w;
    o.h[4]=(__bf16)f1.x; o.h[5]=(__bf16)f1.y; o.h[6]=(__bf16)f1.z; o.h[7]=(__bf16)f1.w;
    *(uint4*)dst = o.q;
}

// ---------------------------------------------------------------
// Kernel 2: V transpose + bf16 convert: Vt[b][h][s] = bf16(V[b][s][h]).
// LDS 32x33 tile to keep both sides coalesced.
// ---------------------------------------------------------------
__global__ void pa_vt_kernel(const float* __restrict__ V,
                             unsigned short* __restrict__ Vt) {
    __shared__ unsigned short tile[32][33];
    int tx = threadIdx.x & 31, ty = threadIdx.x >> 5;   // 32x8
    int blk = blockIdx.x;
    int hTile = blk % (H / 32); blk /= (H / 32);
    int sTile = blk % (S / 32); blk /= (S / 32);
    int b = blk;
    const float* Vb = V + (size_t)b * S * H;
    unsigned short* Vtb = Vt + (size_t)b * H * S;
    int h0 = hTile * 32, s0 = sTile * 32;
    #pragma unroll
    for (int r = 0; r < 32; r += 8)
        tile[ty + r][tx] = bfbits(Vb[(size_t)(s0 + ty + r) * H + h0 + tx]);
    __syncthreads();
    #pragma unroll
    for (int r = 0; r < 32; r += 8)
        Vtb[(size_t)(h0 + ty + r) * S + s0 + tx] = tile[tx][ty + r];
}

// ---------------------------------------------------------------
// Kernel 3: rfft-phase features via WMMA. One wave = 64 rows x 16 freqs:
// 4 A-frags reuse each (cos,sin) basis fragment pair -> 8 WMMA / k-step.
// feature = (re, -im_acc) / |z|; f >= NF zero-padded.
// ---------------------------------------------------------------
__global__ void pa_feat_kernel(const unsigned short* __restrict__ Qh,
                               const unsigned short* __restrict__ Kh,
                               const unsigned short* __restrict__ cosBT,
                               const unsigned short* __restrict__ sinBT,
                               unsigned short* __restrict__ featQ,
                               unsigned short* __restrict__ featK) {
    constexpr int FT = FP / 16;    // 33
    constexpr int IT = S  / 64;    // 32 (64-row tiles)
    int wid = blockIdx.x * WPB + (threadIdx.x >> 5);
    int fTile = wid % FT; wid /= FT;
    int iTile = wid % IT; wid /= IT;
    int b     = wid % B;  wid /= B;
    int t     = wid;               // 0 = Q, 1 = K

    const unsigned short* X   = (t == 0 ? Qh : Kh) + (size_t)b * S * H;
    unsigned short* Feat = (t == 0 ? featQ : featK) + (size_t)b * S * KK;

    int lane = threadIdx.x & 31;
    int hi   = (lane >> 4) & 1;
    int mrow = lane & 15;
    int iBase = iTile * 64;

    const unsigned short* aRow[4];
    #pragma unroll
    for (int r = 0; r < 4; ++r)
        aRow[r] = X + (size_t)(iBase + r * 16 + mrow) * H + hi * 16;
    const unsigned short* bcRow = cosBT + (size_t)(fTile * 16 + mrow) * H + hi * 16;
    const unsigned short* bsRow = sinBT + (size_t)(fTile * 16 + mrow) * H + hi * 16;

    v8f re[4] = {}, im[4] = {};
    for (int k0 = 0; k0 < H; k0 += 32) {
        v16bf bc = load_b_bf16(bcRow + k0);
        v16bf bs = load_b_bf16(bsRow + k0);
        #pragma unroll
        for (int r = 0; r < 4; ++r) {
            v16bf a = load_a_bf16(aRow[r] + k0);
            re[r] = WMMA_BF16(a, bc, re[r]);
            im[r] = WMMA_BF16(a, bs, im[r]);
        }
    }

    int f = fTile * 16 + mrow;     // C/D: N = lane&15
    #pragma unroll
    for (int r = 0; r < 4; ++r) {
        #pragma unroll
        for (int v = 0; v < 8; ++v) {
            int row = iBase + r * 16 + v + 8 * hi;      // C/D: M = v + 8*hi
            float rv = re[r][v];
            float iv = -im[r][v];                       // rfft: im = -sum x*sin
            float r2 = rv * rv + iv * iv;
            float c, s;
            if (f >= NF)       { c = 0.f; s = 0.f; }
            else if (r2 > 0.f) { float ri = rsqrtf(r2); c = rv * ri; s = iv * ri; }
            else               { c = 1.f; s = 0.f; }    // atan2(0,0)=0
            unsigned short* fr = Feat + (size_t)row * KK;
            fr[f]      = bfbits(c);
            fr[FP + f] = bfbits(s);
        }
    }
}

// ---------------------------------------------------------------
// Kernel 4: scores = (Fq @ Fk^T) / NF  (+1.0 is softmax-invariant).
// One wave = 64x32 score tile: 4x2 register blocking, 8 WMMA / k-step.
// B = Fk^T so B-fragment reads are K-contiguous.
// ---------------------------------------------------------------
__global__ void pa_scores_kernel(const unsigned short* __restrict__ featQ,
                                 const unsigned short* __restrict__ featK,
                                 float* __restrict__ Wout) {
    constexpr int JT = S / 32, IT = S / 64;
    int wid = blockIdx.x * WPB + (threadIdx.x >> 5);
    int jTile = wid % JT; wid /= JT;
    int iTile = wid % IT; wid /= IT;
    int b     = wid;

    const unsigned short* Fq = featQ + (size_t)b * S * KK;
    const unsigned short* Fk = featK + (size_t)b * S * KK;
    float* Wb = Wout + (size_t)b * S * S;

    int lane = threadIdx.x & 31;
    int hi   = (lane >> 4) & 1;
    int mrow = lane & 15;
    int iBase = iTile * 64, jBase = jTile * 32;

    const unsigned short* aRow[4];
    #pragma unroll
    for (int r = 0; r < 4; ++r)
        aRow[r] = Fq + (size_t)(iBase + r * 16 + mrow) * KK + hi * 16;
    const unsigned short* bRow[2];
    #pragma unroll
    for (int u = 0; u < 2; ++u)
        bRow[u] = Fk + (size_t)(jBase + u * 16 + mrow) * KK + hi * 16;

    v8f acc[4][2] = {};
    for (int k0 = 0; k0 < KK; k0 += 32) {
        v16bf b0 = load_b_bf16(bRow[0] + k0);
        v16bf b1 = load_b_bf16(bRow[1] + k0);
        #pragma unroll
        for (int r = 0; r < 4; ++r) {
            v16bf a = load_a_bf16(aRow[r] + k0);
            acc[r][0] = WMMA_BF16(a, b0, acc[r][0]);
            acc[r][1] = WMMA_BF16(a, b1, acc[r][1]);
        }
    }
    const float scale = 1.0f / (float)NF;
    #pragma unroll
    for (int r = 0; r < 4; ++r) {
        #pragma unroll
        for (int v = 0; v < 8; ++v) {
            int row = iBase + r * 16 + v + 8 * hi;
            Wb[(size_t)row * S + jBase + mrow]      = acc[r][0][v] * scale;
            Wb[(size_t)row * S + jBase + 16 + mrow] = acc[r][1][v] * scale;
        }
    }
}

// ---------------------------------------------------------------
// Kernel 5: in-place row softmax over S=2048 (one block per row),
// plus a bf16 copy of the weights for the AV GEMM A-operand.
// ---------------------------------------------------------------
__global__ void pa_softmax_kernel(float* __restrict__ Wt,
                                  unsigned short* __restrict__ Wh) {
    __shared__ float red[256];
    float* row = Wt + (size_t)blockIdx.x * S;
    unsigned short* hrow = Wh + (size_t)blockIdx.x * S;
    int tid = threadIdx.x;

    float m = -3.402823466e+38f;
    for (int j = tid; j < S; j += 256) m = fmaxf(m, row[j]);
    red[tid] = m; __syncthreads();
    for (int s = 128; s > 0; s >>= 1) {
        if (tid < s) red[tid] = fmaxf(red[tid], red[tid + s]);
        __syncthreads();
    }
    m = red[0]; __syncthreads();

    float sum = 0.f;
    for (int j = tid; j < S; j += 256) {
        float e = __expf(row[j] - m);
        row[j] = e;
        sum += e;
    }
    red[tid] = sum; __syncthreads();
    for (int s = 128; s > 0; s >>= 1) {
        if (tid < s) red[tid] += red[tid + s];
        __syncthreads();
    }
    float inv = 1.0f / red[0];
    for (int j = tid; j < S; j += 256) {
        float w = row[j] * inv;
        row[j]  = w;
        hrow[j] = bfbits(w);
    }
}

// ---------------------------------------------------------------
// Kernel 6: output = bf16(weights) @ V using pre-transposed bf16
// Vt[H][S]. One wave = 64 rows x 32 cols, 4x2 blocking, 8 WMMA / k-step.
// ---------------------------------------------------------------
__global__ void pa_av_kernel(const unsigned short* __restrict__ Wh,
                             const unsigned short* __restrict__ Vt,
                             float* __restrict__ Out) {
    constexpr int HT = H / 32, IT = S / 64;
    int wid = blockIdx.x * WPB + (threadIdx.x >> 5);
    int hTile = wid % HT; wid /= HT;
    int iTile = wid % IT; wid /= IT;
    int b     = wid;

    const unsigned short* Wb  = Wh + (size_t)b * S * S;
    const unsigned short* Vtb = Vt + (size_t)b * H * S;
    float* Ob = Out + (size_t)b * S * H;

    int lane = threadIdx.x & 31;
    int hi   = (lane >> 4) & 1;
    int mrow = lane & 15;
    int iBase = iTile * 64, hBase = hTile * 32;

    const unsigned short* aRow[4];
    #pragma unroll
    for (int r = 0; r < 4; ++r)
        aRow[r] = Wb + (size_t)(iBase + r * 16 + mrow) * S + hi * 16;
    const unsigned short* bRow[2];
    #pragma unroll
    for (int u = 0; u < 2; ++u)
        bRow[u] = Vtb + (size_t)(hBase + u * 16 + mrow) * S + hi * 16;

    v8f acc[4][2] = {};
    for (int k0 = 0; k0 < S; k0 += 32) {
        v16bf b0 = load_b_bf16(bRow[0] + k0);
        v16bf b1 = load_b_bf16(bRow[1] + k0);
        #pragma unroll
        for (int r = 0; r < 4; ++r) {
            v16bf a = load_a_bf16(aRow[r] + k0);
            acc[r][0] = WMMA_BF16(a, b0, acc[r][0]);
            acc[r][1] = WMMA_BF16(a, b1, acc[r][1]);
        }
    }
    #pragma unroll
    for (int r = 0; r < 4; ++r) {
        #pragma unroll
        for (int v = 0; v < 8; ++v) {
            int row = iBase + r * 16 + v + 8 * hi;
            Ob[(size_t)row * H + hBase + mrow]      = acc[r][0][v];
            Ob[(size_t)row * H + hBase + 16 + mrow] = acc[r][1][v];
        }
    }
}

// ---------------------------------------------------------------
extern "C" void kernel_launch(void* const* d_in, const int* in_sizes, int n_in,
                              void* d_out, int out_size, void* d_ws, size_t ws_size,
                              hipStream_t stream) {
    const float* Q = (const float*)d_in[0];
    const float* K = (const float*)d_in[1];
    const float* V = (const float*)d_in[2];

    float* outBuf  = (float*)d_out;                 // [B,S,H]
    float* weights = outBuf + (size_t)B * S * H;    // [B,S,S]

    // workspace (bf16 as u16), ~87 MB total:
    //   basisT | featQ | featK | Vt | Qh | Kh
    // Wh (bf16 weights, B*S*S) aliases [Qh|Kh] (dead after feat kernel,
    // and 2*B*S*H == B*S*S exactly).
    unsigned short* cosBT = (unsigned short*)d_ws;
    unsigned short* sinBT = cosBT + (size_t)FP * H;
    unsigned short* featQ = sinBT + (size_t)FP * H;
    unsigned short* featK = featQ + (size_t)B * S * KK;
    unsigned short* Vtb   = featK + (size_t)B * S * KK;
    unsigned short* Qh    = Vtb   + (size_t)B * H * S;
    unsigned short* Kh    = Qh    + (size_t)B * S * H;
    unsigned short* Wh    = Qh;   // alias

    {   // K0: twiddle basis (transposed)
        int total = FP * H;
        pa_basis_kernel<<<(total + 255) / 256, 256, 0, stream>>>(cosBT, sinBT);
    }
    {   // K1: Q,K -> bf16
        size_t threads = (size_t)2 * B * S * H / 8;
        pa_cvt_kernel<<<(int)(threads / 256), 256, 0, stream>>>(Q, K, Qh, Kh);
    }
    {   // K2: V transpose + bf16
        int blocks = B * (S / 32) * (H / 32);
        pa_vt_kernel<<<blocks, 256, 0, stream>>>(V, Vtb);
    }
    {   // K3: DFT + phase features
        int waves = 2 * B * (S / 64) * (FP / 16);
        pa_feat_kernel<<<waves / WPB, 256, 0, stream>>>(Qh, Kh, cosBT, sinBT, featQ, featK);
    }
    {   // K4: scores GEMM into weights output region
        int waves = B * (S / 64) * (S / 32);
        pa_scores_kernel<<<waves / WPB, 256, 0, stream>>>(featQ, featK, weights);
    }
    // K5: row softmax in place + bf16 weights (overwrites dead Qh/Kh)
    pa_softmax_kernel<<<B * S, 256, 0, stream>>>(weights, Wh);
    {   // K6: output = weights @ V
        int waves = B * (S / 64) * (H / 32);
        pa_av_kernel<<<waves / WPB, 256, 0, stream>>>(Wh, Vtb, outBuf);
    }
}
